// SAGPoolModel_25366076850811
// MI455X (gfx1250) — compile-verified, matrix operand order checked by reading, not run on hardware
//
#include <hip/hip_runtime.h>
#include <hip/hip_bf16.h>

// ---------------- problem constants (match reference) ----------------
#define NB      16          // graphs
#define NPG     8192        // nodes per graph
#define NN      (NB * NPG)  // 131072 nodes
#define NE      524288      // edges
#define F_NODE  16
#define DH      128         // hidden dim
#define NL      3           // gnn layers
#define KSEL    4096        // top-k per graph
#define EPSBN   1e-5f

typedef __attribute__((ext_vector_type(16))) __bf16         v16bf;
typedef __attribute__((ext_vector_type(16))) unsigned short v16u;
typedef __attribute__((ext_vector_type(8)))  float          v8f;
typedef __attribute__((ext_vector_type(8)))  unsigned int   u32x8;

// float -> bf16 (round to nearest even), pure integer math
__device__ __forceinline__ unsigned short f2bf(float f) {
    unsigned u = __builtin_bit_cast(unsigned, f);
    unsigned r = u + 0x7FFFu + ((u >> 16) & 1u);
    return (unsigned short)(r >> 16);
}

// orderable key: larger float  <->  larger key
__device__ __forceinline__ unsigned fkey(float f) {
    unsigned u = __builtin_bit_cast(unsigned, f);
    return (u & 0x80000000u) ? ~u : (u | 0x80000000u);
}

// ---------------- 1) node embedding: h = x @ node_w + node_b ----------------
__global__ __launch_bounds__(128) void k_embed(const float* __restrict__ x,
                                               const float* __restrict__ nw,
                                               const float* __restrict__ nb,
                                               float* __restrict__ h) {
    int n = blockIdx.x;            // one node per block
    int c = threadIdx.x;           // 128 cols
    __shared__ float xr[F_NODE];
    if (c < F_NODE) xr[c] = x[(size_t)n * F_NODE + c];
    __syncthreads();
    float acc = nb[c];
#pragma unroll
    for (int k = 0; k < F_NODE; ++k) acc += xr[k] * nw[k * DH + c];
    h[(size_t)n * DH + c] = acc;
}

// ---------------- 2) degree / dis / coef ----------------
__global__ __launch_bounds__(256) void k_deg(const int* __restrict__ dst, float* __restrict__ deg) {
    int e = blockIdx.x * 256 + threadIdx.x;
    if (e < NE) atomicAdd(&deg[dst[e]], 1.0f);
}
__global__ __launch_bounds__(256) void k_dis(float* __restrict__ dis) {
    int n = blockIdx.x * 256 + threadIdx.x;
    if (n < NN) dis[n] = rsqrtf(dis[n] + 1.0f);   // deg = in-degree + 1 (self loop)
}
__global__ __launch_bounds__(256) void k_coef(const int* __restrict__ src, const int* __restrict__ dst,
                                              const float* __restrict__ dis, float* __restrict__ coef) {
    int e = blockIdx.x * 256 + threadIdx.x;
    if (e < NE) coef[e] = dis[src[e]] * dis[dst[e]];
}

// ---------------- 3) swizzle conv weights into fragment-major bf16 ----------------
// layout: Wsw[l][ ((s*4 + kc)*32 + lane)*16 + j ]
//   n = s*16 + (lane&15);  half = lane>>4
//   k = kc*32 + (j<8 ? half*8 + j : 16 + half*8 + (j-8))
__global__ __launch_bounds__(256) void k_swzw(const float* __restrict__ w, unsigned short* __restrict__ wsw) {
    int i = blockIdx.x * 256 + threadIdx.x;
    if (i >= NL * DH * DH) return;
    int l    = i >> 14;
    int r    = i & 16383;
    int j    = r & 15;
    int lane = (r >> 4) & 31;
    int kc   = (r >> 9) & 3;
    int s    = (r >> 11) & 7;
    int half = lane >> 4;
    int n    = s * 16 + (lane & 15);
    int k    = kc * 32 + ((j < 8) ? (half * 8 + j) : (16 + half * 8 + (j - 8)));
    wsw[i] = f2bf(w[l * DH * DH + k * DH + n]);
}

// ---------------- 4) WMMA GEMM: C[M x 128] = A[M x 128] * W[128 x 128] ----------------
// block = 256 threads (8 waves), computes 64 rows x 128 cols.
// wave w owns col stripe n0=w*16; B frags come straight from L2 (pre-swizzled, 32B/lane);
// A strip is staged fp32->bf16 into LDS in fragment-major order (32B/lane reads).
__global__ __launch_bounds__(256) void k_gemm(const float* __restrict__ A,
                                              const unsigned short* __restrict__ Wsw,
                                              float* __restrict__ C, int nrows) {
    // As[rt][kc][lane][j] : 16*... = 4*4*32*16 bf16 = 16 KB
    __shared__ unsigned short As[4 * 4 * 32 * 16];
    const int tid = threadIdx.x;
    const int m0  = blockIdx.x * 64;

    // stage A strip (64x128 fp32 -> bf16, swizzled): 2048 float4 / 256 threads
    {
        const float4* s = (const float4*)(A + (size_t)m0 * DH);
#pragma unroll
        for (int i = 0; i < 8; ++i) {
            int e4 = tid + 256 * i;
            float4 v = s[e4];
            int r  = e4 >> 5;              // row 0..63
            int cf = (e4 & 31) * 4;        // col 0..124 (mult of 4)
            int rt = r >> 4, m = r & 15;
            int kc = cf >> 5, cc = cf & 31;
            int half = (cc >> 3) & 1;
            int j    = (cc & 7) + ((cc >> 4) << 3);   // 4 consecutive j's
            int lane = half * 16 + m;
            unsigned short* p = &As[(((rt * 4 + kc) * 32 + lane) << 4) + j];
            p[0] = f2bf(v.x); p[1] = f2bf(v.y); p[2] = f2bf(v.z); p[3] = f2bf(v.w);
        }
    }
    if (m0 + 64 < nrows)                   // speculative prefetch of next strip
        __builtin_prefetch(A + (size_t)(m0 + 64) * DH, 0, 1);
    __syncthreads();

    const int wave  = tid >> 5;
    const int lane  = tid & 31;
    const int n0    = wave * 16;
    const int lr    = lane & 15;           // N for B,C-frag
    const int rowad = (lane >> 4) * 8;     // C-frag row add

    // preload the 4 B fragments for this wave's stripe (32B contiguous per lane)
    v16bf bfr[4];
#pragma unroll
    for (int kc = 0; kc < 4; ++kc) {
        u32x8 bv = *(const u32x8*)(Wsw + ((((wave * 4 + kc) * 32) + lane) << 4));
        bfr[kc] = __builtin_bit_cast(v16bf, bv);
    }

    v8f acc[4] = {};
#pragma unroll
    for (int kc = 0; kc < 4; ++kc) {
#pragma unroll
        for (int rt = 0; rt < 4; ++rt) {
            u32x8 av = *(const u32x8*)&As[(((rt * 4 + kc) * 32) + lane) << 4];
            v16bf af = __builtin_bit_cast(v16bf, av);
            acc[rt] = __builtin_amdgcn_wmma_f32_16x16x32_bf16(
                false, af, false, bfr[kc], (short)0, acc[rt], false, false);
        }
    }
#pragma unroll
    for (int rt = 0; rt < 4; ++rt)
#pragma unroll
        for (int r = 0; r < 8; ++r)
            C[(size_t)(m0 + rt * 16 + r + rowad) * DH + n0 + lr] = acc[rt][r];
}

// ---------------- 5) edge scatter: out[dst] += feat[src] * coef (coef==null -> 1) ----------------
__global__ __launch_bounds__(256) void k_scatter(const float* __restrict__ feat,
                                                 float* __restrict__ out,
                                                 const int* __restrict__ src,
                                                 const int* __restrict__ dst,
                                                 const float* __restrict__ coef) {
    int gw   = (blockIdx.x * 256 + threadIdx.x) >> 5;   // one wave per edge
    int lane = threadIdx.x & 31;
    if (gw >= NE) return;
    int s = src[gw], d = dst[gw];
    float c = coef ? coef[gw] : 1.0f;
    float4 v = ((const float4*)(feat + (size_t)s * DH))[lane];
    float* dp = out + (size_t)d * DH + lane * 4;
    atomicAdd(dp + 0, v.x * c);
    atomicAdd(dp + 1, v.y * c);
    atomicAdd(dp + 2, v.z * c);
    atomicAdd(dp + 3, v.w * c);
}

// ---------------- 6) h = leaky(BN(agg + hw*dis^2 + conv_b)) ----------------
__global__ __launch_bounds__(256) void k_update(const float* __restrict__ agg,
                                                const float* __restrict__ hw,
                                                const float* __restrict__ dis,
                                                const float* __restrict__ cb,
                                                const float* __restrict__ bg,
                                                const float* __restrict__ bb,
                                                const float* __restrict__ bm,
                                                const float* __restrict__ bv,
                                                float* __restrict__ h) {
    int i = blockIdx.x * 256 + threadIdx.x;
    if (i >= NN * DH) return;
    int n = i >> 7, c = i & 127;
    float d2 = dis[n]; d2 *= d2;
    float v  = agg[i] + hw[i] * d2 + cb[c];
    v = (v - bm[c]) * rsqrtf(bv[c] + EPSBN) * bg[c] + bb[c];
    h[i] = v > 0.0f ? v : 0.01f * v;
}

// ---------------- 7) score = nagg@wr + h@ws + b ----------------
__global__ __launch_bounds__(256) void k_score(const float* __restrict__ nagg,
                                               const float* __restrict__ h,
                                               const float* __restrict__ wr,
                                               const float* __restrict__ wsv,
                                               const float* __restrict__ pb,
                                               float* __restrict__ score) {
    int n    = (blockIdx.x * 256 + threadIdx.x) >> 5;  // one wave per node
    int lane = threadIdx.x & 31;
    if (n >= NN) return;
    float4 a = ((const float4*)(nagg + (size_t)n * DH))[lane];
    float4 b = ((const float4*)(h    + (size_t)n * DH))[lane];
    float4 r = ((const float4*)wr)[lane];
    float4 s = ((const float4*)wsv)[lane];
    float p = a.x * r.x + a.y * r.y + a.z * r.z + a.w * r.w
            + b.x * s.x + b.y * s.y + b.z * s.z + b.w * s.w;
#pragma unroll
    for (int m = 16; m >= 1; m >>= 1) p += __shfl_xor(p, m, 32);
    if (lane == 0) score[n] = p + pb[0];
}

// ---------------- 8) per-graph radix select of K-th largest ----------------
__global__ __launch_bounds__(256) void k_select(const float* __restrict__ score,
                                                unsigned* __restrict__ thresh,
                                                unsigned* __restrict__ needeq,
                                                unsigned* __restrict__ eqctr) {
    int g = blockIdx.x;
    const float* s = score + (size_t)g * NPG;
    __shared__ unsigned hist[16];
    __shared__ unsigned sprefix, srem;
    if (threadIdx.x == 0) { sprefix = 0u; srem = KSEL; }
    __syncthreads();
    for (int shift = 28; shift >= 0; shift -= 4) {
        if (threadIdx.x < 16) hist[threadIdx.x] = 0u;
        __syncthreads();
        unsigned prefix = sprefix;
        unsigned himask = (shift == 28) ? 0u : (0xFFFFFFFFu << (shift + 4));
        for (int i = threadIdx.x; i < NPG; i += 256) {
            unsigned k = fkey(s[i]);
            if ((k & himask) == prefix) atomicAdd(&hist[(k >> shift) & 15], 1u);
        }
        __syncthreads();
        if (threadIdx.x == 0) {
            unsigned rem = srem;
            int b = 15;
            for (; b > 0; --b) { if (hist[b] >= rem) break; rem -= hist[b]; }
            sprefix = prefix | ((unsigned)b << shift);
            srem    = rem;
        }
        __syncthreads();
    }
    if (threadIdx.x == 0) {
        thresh[g] = sprefix;   // key of K-th largest
        needeq[g] = srem;      // #equal-to-threshold to admit
        eqctr[g]  = 0u;
    }
}

// ---------------- 9) pooled sum: xg[g] += h[n] * tanh(score[n]) over selected ----------------
__global__ __launch_bounds__(256) void k_pool(const float* __restrict__ h,
                                              const float* __restrict__ score,
                                              const unsigned* __restrict__ thresh,
                                              const unsigned* __restrict__ needeq,
                                              unsigned* __restrict__ eqctr,
                                              float* __restrict__ xg) {
    int n    = (blockIdx.x * 256 + threadIdx.x) >> 5;  // one wave per node
    int lane = threadIdx.x & 31;
    if (n >= NN) return;
    int g = n >> 13;                                   // NPG = 8192
    float sc = score[n];
    unsigned k  = fkey(sc);
    unsigned tk = thresh[g];
    bool take = k > tk;
    if (k == tk) {
        int tick = 0;
        if (lane == 0) tick = (int)atomicAdd(&eqctr[g], 1u);
        tick = __builtin_amdgcn_readfirstlane(tick);
        take = (unsigned)tick < needeq[g];
    }
    if (take) {
        float w = tanhf(sc);
        float4 v = ((const float4*)(h + (size_t)n * DH))[lane];
        float* dp = xg + (size_t)g * DH + lane * 4;
        atomicAdd(dp + 0, v.x * w);
        atomicAdd(dp + 1, v.y * w);
        atomicAdd(dp + 2, v.z * w);
        atomicAdd(dp + 3, v.w * w);
    }
}

// ---------------- 10) readout MLP: out = leaky(xg@r1+b1)@r2+b2 ----------------
__global__ __launch_bounds__(256) void k_mlp(const float* __restrict__ xg,
                                             const float* __restrict__ r1w, const float* __restrict__ r1b,
                                             const float* __restrict__ r2w, const float* __restrict__ r2b,
                                             float* __restrict__ out) {
    __shared__ float t1[NB * 64];
    int tid = threadIdx.x;
    for (int e = tid; e < NB * 64; e += 256) {
        int b = e >> 6, o = e & 63;
        float a = r1b[o];
        for (int k = 0; k < DH; ++k) a += xg[b * DH + k] * r1w[k * 64 + o];
        t1[e] = a > 0.0f ? a : 0.01f * a;
    }
    __syncthreads();
    if (tid < NB * 4) {
        int b = tid >> 2, o = tid & 3;
        float a = r2b[o];
        for (int k = 0; k < 64; ++k) a += t1[b * 64 + k] * r2w[k * 4 + o];
        out[b * 4 + o] = a;
    }
}

// ============================ launcher ============================
extern "C" void kernel_launch(void* const* d_in, const int* in_sizes, int n_in,
                              void* d_out, int out_size, void* d_ws, size_t ws_size,
                              hipStream_t stream) {
    const float* x      = (const float*)d_in[0];
    const int*   ei     = (const int*)  d_in[2];
    const int*   src    = ei;
    const int*   dst    = ei + NE;
    const float* node_w = (const float*)d_in[3];
    const float* node_b = (const float*)d_in[4];
    const float* conv_w = (const float*)d_in[7];
    const float* conv_b = (const float*)d_in[8];
    const float* bn_g   = (const float*)d_in[9];
    const float* bn_b   = (const float*)d_in[10];
    const float* bn_m   = (const float*)d_in[11];
    const float* bn_v   = (const float*)d_in[12];
    const float* p_wr   = (const float*)d_in[13];
    const float* p_ws   = (const float*)d_in[14];
    const float* p_b    = (const float*)d_in[15];
    const float* r1w    = (const float*)d_in[16];
    const float* r1b    = (const float*)d_in[17];
    const float* r2w    = (const float*)d_in[18];
    const float* r2b    = (const float*)d_in[19];

    // ---- workspace carve (256B aligned) ----
    char* base = (char*)d_ws;
    size_t off = 0;
    auto carve = [&](size_t bytes) -> void* {
        void* p = base + off;
        off = (off + bytes + 255) & ~(size_t)255;
        return p;
    };
    float*          h     = (float*)carve((size_t)NN * DH * 4);
    float*          hw    = (float*)carve((size_t)NN * DH * 4);
    float*          agg   = (float*)carve((size_t)NN * DH * 4);
    float*          dis   = (float*)carve((size_t)NN * 4);
    float*          coef  = (float*)carve((size_t)NE * 4);
    float*          score = (float*)carve((size_t)NN * 4);
    unsigned short* Wsw   = (unsigned short*)carve((size_t)NL * DH * DH * 2);
    float*          xg    = (float*)carve((size_t)NB * DH * 4);
    unsigned*       thr   = (unsigned*)carve(NB * 4);
    unsigned*       neq   = (unsigned*)carve(NB * 4);
    unsigned*       eqc   = (unsigned*)carve(NB * 4);

    // 1) embed
    k_embed<<<NN, 128, 0, stream>>>(x, node_w, node_b, h);

    // 2) gcn_norm
    hipMemsetAsync(dis, 0, (size_t)NN * 4, stream);
    k_deg <<<NE / 256, 256, 0, stream>>>(dst, dis);
    k_dis <<<NN / 256, 256, 0, stream>>>(dis);
    k_coef<<<NE / 256, 256, 0, stream>>>(src, dst, dis, coef);

    // 3) weights -> swizzled bf16 fragments
    k_swzw<<<(NL * DH * DH) / 256, 256, 0, stream>>>(conv_w, Wsw);

    // 4) GNN layers
    for (int l = 0; l < NL; ++l) {
        k_gemm<<<NN / 64, 256, 0, stream>>>(h, Wsw + (size_t)l * DH * DH, hw, NN);
        hipMemsetAsync(agg, 0, (size_t)NN * DH * 4, stream);
        k_scatter<<<NE / 8, 256, 0, stream>>>(hw, agg, src, dst, coef);
        k_update<<<(NN * DH) / 256, 256, 0, stream>>>(agg, hw, dis,
            conv_b + l * DH, bn_g + l * DH, bn_b + l * DH, bn_m + l * DH, bn_v + l * DH, h);
    }

    // 5) SAGPooling score
    hipMemsetAsync(agg, 0, (size_t)NN * DH * 4, stream);
    k_scatter<<<NE / 8, 256, 0, stream>>>(h, agg, src, dst, (const float*)nullptr);
    k_score<<<NN / 8, 256, 0, stream>>>(agg, h, p_wr, p_ws, p_b, score);

    // 6) per-graph top-K threshold + pooled sum
    k_select<<<NB, 256, 0, stream>>>(score, thr, neq, eqc);
    hipMemsetAsync(xg, 0, (size_t)NB * DH * 4, stream);
    k_pool<<<NN / 8, 256, 0, stream>>>(h, score, thr, neq, eqc, xg);

    // 7) readout
    k_mlp<<<1, 256, 0, stream>>>(xg, r1w, r1b, r2w, r2b, (float*)d_out);
}
